// AttnBlock_13151189860490
// MI455X (gfx1250) — compile-verified, matrix operand order checked by reading, not run on hardware
//
#include <hip/hip_runtime.h>
#include <hip/hip_bf16.h>

// ---------------------------------------------------------------------------
// AttnBlock for MI455X (gfx1250): GroupNorm -> q,k,v nin -> full attention
// (flash-style online softmax, cooperative QK^T) -> out nin -> +residual.
// GEMMs via v_wmma_f32_16x16x32_f16 (wave32); weight tiles staged in LDS via
// global_load_async_to_lds_b128 (ASYNCcnt) when the toolchain exposes it.
// ---------------------------------------------------------------------------

#define BATCH 32
#define CDIM  512
#define NTOK  1024          // H*W = 32*32
#define NGRP  32
#define GCH   16            // CDIM / NGRP
#define QKSCL 0.044194173824159216f   // 1/sqrt(512)

typedef _Float16 f16x16 __attribute__((ext_vector_type(16)));
typedef _Float16 f16x8  __attribute__((ext_vector_type(8)));
typedef float    f32x8  __attribute__((ext_vector_type(8)));
typedef int      v4i    __attribute__((ext_vector_type(4)));

#if __has_builtin(__builtin_amdgcn_global_load_async_to_lds_b128) && \
    __has_builtin(__builtin_amdgcn_s_wait_asynccnt)
#define HAVE_ASYNC_LDS 1
#define AS1G __attribute__((address_space(1)))
#define AS3L __attribute__((address_space(3)))
#else
#define HAVE_ASYNC_LDS 0
#endif

// Load a 16x32 f16 fragment in CDNA5 A/B layout from a row-major matrix with
// leading dimension ld (halfs). "Row" for A = M index, for B = N index (NT).
// Layout (ISA 7.12.2, 16-bit A 16x32): lane<16 holds K{0..7,16..23},
// lane>=16 holds K{8..15,24..31}, 2 halfs per VGPR.
__device__ __forceinline__ f16x16 load_frag(const _Float16* __restrict__ p, int ld) {
    int lane = threadIdx.x & 31;
    int r    = lane & 15;
    int hi   = lane >> 4;
    const _Float16* q = p + (size_t)r * (size_t)ld + hi * 8;
    f16x8 lo = *reinterpret_cast<const f16x8*>(q);
    f16x8 hh = *reinterpret_cast<const f16x8*>(q + 16);
    f16x16 f;
#pragma unroll
    for (int i = 0; i < 8; ++i) { f[i] = lo[i]; f[i + 8] = hh[i]; }
    return f;
}

__device__ __forceinline__ f32x8 wmma_f16(f16x16 a, f16x16 b, f32x8 c) {
    return __builtin_amdgcn_wmma_f32_16x16x32_f16(false, a, false, b,
                                                  (short)0, c, false, false);
}

// Reductions across the 16 lanes of a half-wave (rows are split lane<16 /
// lane>=16 in the C/D fragment layout; xor masks < 16 never cross halves).
__device__ __forceinline__ float rowmax16(float v) {
    v = fmaxf(v, __shfl_xor(v, 1, 32));
    v = fmaxf(v, __shfl_xor(v, 2, 32));
    v = fmaxf(v, __shfl_xor(v, 4, 32));
    v = fmaxf(v, __shfl_xor(v, 8, 32));
    return v;
}
__device__ __forceinline__ float rowsum16(float v) {
    v += __shfl_xor(v, 1, 32);
    v += __shfl_xor(v, 2, 32);
    v += __shfl_xor(v, 4, 32);
    v += __shfl_xor(v, 8, 32);
    return v;
}

// ---------------------------------------------------------------------------
// Cooperative staging of a 128(rows) x 32(k) f16 B-tile (8KB) into LDS.
// 256 threads, 2 x 16B chunks each. Rows strided by CDIM in global.
// ---------------------------------------------------------------------------
__device__ __forceinline__ void stage_b_tile(const _Float16* __restrict__ Bg,
                                             int k0, _Float16* lB) {
    int t = threadIdx.x;            // 0..255
#pragma unroll
    for (int i = 0; i < 2; ++i) {
        int chunk = t + i * 256;    // 0..511
        int row   = chunk >> 2;     // 0..127
        int off   = (chunk & 3) * 8;
        const _Float16* g = Bg + (size_t)row * CDIM + k0 + off;
        _Float16*       l = lB + row * 32 + off;
#if HAVE_ASYNC_LDS
        __builtin_amdgcn_global_load_async_to_lds_b128(
            (AS1G v4i*)g, (AS3L v4i*)l, 0, 0);
#else
        *reinterpret_cast<f16x8*>(l) = *reinterpret_cast<const f16x8*>(g);
#endif
    }
}

__device__ __forceinline__ void stage_wait() {
#if HAVE_ASYNC_LDS
    __builtin_amdgcn_s_wait_asynccnt(0);
#endif
}

// Shared 16x128 per-wave tile mainloop: acc += A(16xK) * B(Kx128), K = CDIM.
// sB = double-buffered LDS tile [2][128*32]. Called by all 256 threads.
__device__ __forceinline__ void gemm_tile_16x128(const _Float16* __restrict__ A0,
                                                 const _Float16* __restrict__ Bg,
                                                 _Float16* sB, f32x8 acc[8]) {
    stage_b_tile(Bg, 0, sB);
    stage_wait();
    __syncthreads();
    for (int k0 = 0; k0 < CDIM; k0 += 32) {
        int cur = (k0 >> 5) & 1;
        if (k0 + 32 < CDIM)
            stage_b_tile(Bg, k0 + 32, sB + (cur ^ 1) * (128 * 32));
        f16x16 a = load_frag(A0 + k0, CDIM);
        const _Float16* lb = sB + cur * (128 * 32);
#pragma unroll
        for (int nb = 0; nb < 8; ++nb) {
            f16x16 bf = load_frag(lb + nb * 16 * 32, 32);
            acc[nb] = wmma_f16(a, bf, acc[nb]);
        }
        stage_wait();
        __syncthreads();
    }
}

// ---------------------------------------------------------------------------
// Kernel 1: W [C,D] f32 -> WT [D,C] f16 for all four projection weights.
// ---------------------------------------------------------------------------
__global__ void transpose_w_kernel(const float* __restrict__ Wq,
                                   const float* __restrict__ Wk,
                                   const float* __restrict__ Wv,
                                   const float* __restrict__ Wt,
                                   _Float16* __restrict__ WT_all) {
    int d = blockIdx.x;
    int w = blockIdx.y;
    const float* src = (w == 0) ? Wq : (w == 1) ? Wk : (w == 2) ? Wv : Wt;
    _Float16* dst = WT_all + (size_t)w * CDIM * CDIM + (size_t)d * CDIM;
    for (int c = threadIdx.x; c < CDIM; c += blockDim.x)
        dst[c] = (_Float16)src[(size_t)c * CDIM + d];
}

// ---------------------------------------------------------------------------
// Kernel 2: GroupNorm; x [B,C,N] f32 -> h16 [B,N,C] f16 (token-major).
// ---------------------------------------------------------------------------
__global__ __launch_bounds__(256)
void groupnorm_kernel(const float* __restrict__ x,
                      const float* __restrict__ gscale,
                      const float* __restrict__ gbias,
                      _Float16* __restrict__ h16) {
    int b = blockIdx.x >> 5;
    int g = blockIdx.x & 31;
    const float* xp = x + ((size_t)b * CDIM + (size_t)g * GCH) * NTOK;
    const int n = GCH * NTOK;

    float s = 0.f, ss = 0.f;
    for (int i = threadIdx.x; i < n; i += 256) {
        float v = xp[i];
        s += v; ss += v * v;
    }
    __shared__ float rs[256], rq[256];
    __shared__ float mv[2];
    rs[threadIdx.x] = s; rq[threadIdx.x] = ss;
    __syncthreads();
#pragma unroll
    for (int off = 128; off > 0; off >>= 1) {
        if ((int)threadIdx.x < off) {
            rs[threadIdx.x] += rs[threadIdx.x + off];
            rq[threadIdx.x] += rq[threadIdx.x + off];
        }
        __syncthreads();
    }
    if (threadIdx.x == 0) {
        float mean = rs[0] / (float)n;
        float var  = rq[0] / (float)n - mean * mean;
        mv[0] = mean;
        mv[1] = __frsqrt_rn(var + 1e-6f);
    }
    __syncthreads();
    float mean = mv[0], inv = mv[1];

    _Float16* hb = h16 + (size_t)b * NTOK * CDIM;
    for (int i = threadIdx.x; i < n; i += 256) {
        int cl = i >> 10;
        int t  = i & (NTOK - 1);
        int c  = g * GCH + cl;
        float v = (xp[i] - mean) * inv * gscale[c] + gbias[c];
        hb[(size_t)t * CDIM + c] = (_Float16)v;
    }
}

// ---------------------------------------------------------------------------
// Kernel 3: q/k/v = h @ W + b. Block = 128 tokens x 128 channels, 8 waves
// each owning 16 tokens; shared LDS-staged weight tile.
// ---------------------------------------------------------------------------
__global__ __launch_bounds__(256)
void qkv_gemm_kernel(const _Float16* __restrict__ h16,
                     const _Float16* __restrict__ WT_all,
                     const float* __restrict__ bq,
                     const float* __restrict__ bk,
                     const float* __restrict__ bv,
                     _Float16* __restrict__ q16,
                     _Float16* __restrict__ k16,
                     _Float16* __restrict__ vT16) {
    __shared__ __align__(16) _Float16 sB[2][128 * 32];

    int w     = blockIdx.z;
    int b     = blockIdx.y;
    int ttile = blockIdx.x >> 2;
    int ctile = blockIdx.x & 3;
    int wave  = threadIdx.x >> 5;
    int lane  = threadIdx.x & 31;
    int t0    = ttile * 128 + wave * 16;
    int c0    = ctile * 128;

    const _Float16* WT   = WT_all + (size_t)w * CDIM * CDIM;
    const float*    bias = (w == 0) ? bq : (w == 1) ? bk : bv;
    const _Float16* A0   = h16 + (size_t)b * NTOK * CDIM + (size_t)t0 * CDIM;

    f32x8 acc[8];
#pragma unroll
    for (int nb = 0; nb < 8; ++nb) acc[nb] = (f32x8)(0.f);

    gemm_tile_16x128(A0, WT + (size_t)c0 * CDIM, &sB[0][0], acc);

    int col_l = lane & 15, hi = lane >> 4;
    _Float16* qb = q16  + (size_t)b * NTOK * CDIM;
    _Float16* kb = k16  + (size_t)b * NTOK * CDIM;
    _Float16* vb = vT16 + (size_t)b * CDIM * NTOK;
#pragma unroll
    for (int nb = 0; nb < 8; ++nb) {
        int c = c0 + nb * 16 + col_l;
        float bs = bias[c];
        if (w == 2) {
            // v stored channel-major: 8 consecutive tokens -> vector store
            f16x8 pk;
#pragma unroll
            for (int j = 0; j < 8; ++j) pk[j] = (_Float16)(acc[nb][j] + bs);
            *reinterpret_cast<f16x8*>(vb + (size_t)c * NTOK + t0 + 8 * hi) = pk;
        } else {
#pragma unroll
            for (int j = 0; j < 8; ++j) {
                int row = t0 + j + 8 * hi;
                float v = acc[nb][j] + bs;
                if (w == 0) qb[(size_t)row * CDIM + c] = (_Float16)(v * QKSCL);
                else        kb[(size_t)row * CDIM + c] = (_Float16)v;
            }
        }
    }
}

// ---------------------------------------------------------------------------
// Kernel 4: flash attention, cooperative QK^T. Block = 4 waves x 16 queries.
// Each wave computes partial S over 128 channels; partials reduced via LDS
// ([row][col][wave] so the reduce is one ds_load_b128 + hadd). All waves run
// the identical online-softmax update, then PV over their own 128 channels.
// ---------------------------------------------------------------------------
__global__ __launch_bounds__(128)
void attn_kernel(const _Float16* __restrict__ q16,
                 const _Float16* __restrict__ k16,
                 const _Float16* __restrict__ vT16,
                 _Float16* __restrict__ h2) {
    __shared__ __align__(16) float    sS[16 * 32 * 4];   // [row][col0..31][wave]
    __shared__ __align__(16) _Float16 sP[4][16 * 32];    // per-wave P staging

    int b     = blockIdx.y;
    int t0    = blockIdx.x * 16;
    int wave  = threadIdx.x >> 5;
    int lane  = threadIdx.x & 31;
    int c0    = wave * 128;       // this wave's output-channel chunk & K chunk
    int col_l = lane & 15, hi = lane >> 4;

    const _Float16* qb = q16 + (size_t)b * NTOK * CDIM + (size_t)t0 * CDIM;
    const _Float16* kb = k16 + (size_t)b * NTOK * CDIM;
    const _Float16* vb = vT16 + (size_t)b * CDIM * NTOK;
    _Float16* pw = &sP[wave][0];

    f32x8 O[8];
#pragma unroll
    for (int nc = 0; nc < 8; ++nc) O[nc] = (f32x8)(0.f);
    float mrun[8], lrun[8];
#pragma unroll
    for (int j = 0; j < 8; ++j) { mrun[j] = -3.0e38f; lrun[j] = 0.f; }

    for (int s0 = 0; s0 < NTOK; s0 += 32) {
        // ---- partial S over this wave's 128 channels (two 16-key subtiles)
        f32x8 Sa = (f32x8)(0.f), Sb = (f32x8)(0.f);
        for (int kk = 0; kk < 128; kk += 32) {
            f16x16 a  = load_frag(qb + c0 + kk, CDIM);
            f16x16 ba = load_frag(kb + (size_t)s0 * CDIM + c0 + kk, CDIM);
            f16x16 bb = load_frag(kb + (size_t)(s0 + 16) * CDIM + c0 + kk, CDIM);
            Sa = wmma_f16(a, ba, Sa);
            Sb = wmma_f16(a, bb, Sb);
        }
        __syncthreads();   // previous iteration's sS reads complete
#pragma unroll
        for (int j = 0; j < 8; ++j) {
            int row = j + 8 * hi;
            sS[(row * 32 + col_l) * 4 + wave]      = Sa[j];
            sS[(row * 32 + 16 + col_l) * 4 + wave] = Sb[j];
        }
        __syncthreads();   // partials visible to all waves

        // ---- reduce partials + online softmax (identical across waves) ----
#pragma unroll
        for (int j = 0; j < 8; ++j) {
            int row = j + 8 * hi;
            float4 va = *reinterpret_cast<const float4*>(&sS[(row * 32 + col_l) * 4]);
            float4 vb4 = *reinterpret_cast<const float4*>(&sS[(row * 32 + 16 + col_l) * 4]);
            float sa = (va.x + va.y) + (va.z + va.w);
            float sb = (vb4.x + vb4.y) + (vb4.z + vb4.w);

            float rmax = rowmax16(fmaxf(sa, sb));
            float mnew = fmaxf(mrun[j], rmax);
            float f    = __expf(mrun[j] - mnew);
            float pa   = __expf(sa - mnew);
            float pb   = __expf(sb - mnew);
            float rsum = rowsum16(pa + pb);
            lrun[j] = lrun[j] * f + rsum;
            mrun[j] = mnew;
#pragma unroll
            for (int nc = 0; nc < 8; ++nc) O[nc][j] *= f;
            pw[row * 32 + col_l]      = (_Float16)pa;   // keys s0..s0+15
            pw[row * 32 + 16 + col_l] = (_Float16)pb;   // keys s0+16..s0+31
        }
        // per-wave LDS ordering (store -> frag load)
        asm volatile("s_wait_dscnt 0" ::: "memory");

        // ---- P (A-layout) x V (NT from channel-major v) ----
        f16x16 pf = load_frag(pw, 32);
#pragma unroll
        for (int nc = 0; nc < 8; ++nc) {
            f16x16 bf = load_frag(vb + (size_t)(c0 + nc * 16) * NTOK + s0, NTOK);
            O[nc] = wmma_f16(pf, bf, O[nc]);
        }
    }

    _Float16* ob = h2 + (size_t)b * NTOK * CDIM;
#pragma unroll
    for (int j = 0; j < 8; ++j) {
        float inv = 1.0f / lrun[j];
        int row = t0 + j + 8 * hi;
#pragma unroll
        for (int nc = 0; nc < 8; ++nc)
            ob[(size_t)row * CDIM + c0 + nc * 16 + col_l] = (_Float16)(O[nc][j] * inv);
    }
}

// ---------------------------------------------------------------------------
// Kernel 5: out = h2 @ Wt + bt + x, f32 [B,C,H,W].
// ---------------------------------------------------------------------------
__global__ __launch_bounds__(256)
void out_proj_kernel(const _Float16* __restrict__ h2,
                     const _Float16* __restrict__ WtT,
                     const float* __restrict__ bt,
                     const float* __restrict__ x,
                     float* __restrict__ out) {
    __shared__ __align__(16) _Float16 sB[2][128 * 32];

    int b     = blockIdx.y;
    int ttile = blockIdx.x >> 2;
    int ctile = blockIdx.x & 3;
    int wave  = threadIdx.x >> 5;
    int lane  = threadIdx.x & 31;
    int t0    = ttile * 128 + wave * 16;
    int c0    = ctile * 128;

    const _Float16* A0 = h2 + (size_t)b * NTOK * CDIM + (size_t)t0 * CDIM;

    f32x8 acc[8];
#pragma unroll
    for (int nb = 0; nb < 8; ++nb) acc[nb] = (f32x8)(0.f);

    gemm_tile_16x128(A0, WtT + (size_t)c0 * CDIM, &sB[0][0], acc);

    int col_l = lane & 15, hi = lane >> 4;
#pragma unroll
    for (int nb = 0; nb < 8; ++nb) {
        int c = c0 + nb * 16 + col_l;
        float bs = bt[c];
        const float* xr  = x   + (size_t)b * CDIM * NTOK + (size_t)c * NTOK + t0 + 8 * hi;
        float*       orr = out + (size_t)b * CDIM * NTOK + (size_t)c * NTOK + t0 + 8 * hi;
        float4 xlo = *reinterpret_cast<const float4*>(xr);
        float4 xhi = *reinterpret_cast<const float4*>(xr + 4);
        float4 olo = make_float4(acc[nb][0] + bs + xlo.x, acc[nb][1] + bs + xlo.y,
                                 acc[nb][2] + bs + xlo.z, acc[nb][3] + bs + xlo.w);
        float4 ohi = make_float4(acc[nb][4] + bs + xhi.x, acc[nb][5] + bs + xhi.y,
                                 acc[nb][6] + bs + xhi.z, acc[nb][7] + bs + xhi.w);
        *reinterpret_cast<float4*>(orr)     = olo;
        *reinterpret_cast<float4*>(orr + 4) = ohi;
    }
}

// ---------------------------------------------------------------------------
extern "C" void kernel_launch(void* const* d_in, const int* in_sizes, int n_in,
                              void* d_out, int out_size, void* d_ws, size_t ws_size,
                              hipStream_t stream) {
    const float* x        = (const float*)d_in[0];
    const float* gn_scale = (const float*)d_in[1];
    const float* gn_bias  = (const float*)d_in[2];
    const float* Wq = (const float*)d_in[3];  const float* bq = (const float*)d_in[4];
    const float* Wk = (const float*)d_in[5];  const float* bk = (const float*)d_in[6];
    const float* Wv = (const float*)d_in[7];  const float* bv = (const float*)d_in[8];
    const float* Wt = (const float*)d_in[9];  const float* bt = (const float*)d_in[10];
    float* out = (float*)d_out;

    // Workspace layout (f16):
    char* ws = (char*)d_ws;
    const size_t szW  = (size_t)4 * CDIM * CDIM * sizeof(_Float16);          // 2 MiB
    const size_t szT  = (size_t)BATCH * NTOK * CDIM * sizeof(_Float16);      // 32 MiB
    _Float16* WT_all = (_Float16*)ws;                 ws += szW;
    _Float16* h16    = (_Float16*)ws;                 ws += szT;
    _Float16* q16    = (_Float16*)ws;                 ws += szT;
    _Float16* k16    = (_Float16*)ws;                 ws += szT;
    _Float16* vT16   = (_Float16*)ws;                 ws += szT;
    _Float16* h2     = (_Float16*)ws;                 ws += szT;
    (void)ws_size; (void)in_sizes; (void)n_in; (void)out_size;

    transpose_w_kernel<<<dim3(CDIM, 4), 128, 0, stream>>>(Wq, Wk, Wv, Wt, WT_all);
    groupnorm_kernel<<<dim3(BATCH * NGRP), 256, 0, stream>>>(x, gn_scale, gn_bias, h16);
    qkv_gemm_kernel<<<dim3(32, BATCH, 3), 256, 0, stream>>>(h16, WT_all, bq, bk, bv,
                                                            q16, k16, vT16);
    attn_kernel<<<dim3(NTOK / 16, BATCH), 128, 0, stream>>>(q16, k16, vT16, h2);
    out_proj_kernel<<<dim3(32, BATCH), 256, 0, stream>>>(
        h2, WT_all + (size_t)3 * CDIM * CDIM, bt, x, out);
}
